// NeuralFieldTuringMachine_87333864997038
// MI455X (gfx1250) — matrix operand order repaired and checked by reading, not compiled, not run on hardware
//
#include <hip/hip_runtime.h>
#include <math.h>

// ---------------------------------------------------------------------------
// NeuralFieldTuringMachine for MI455X (gfx1250, wave32, WMMA).
//
// write_in @ W decomposed by concat blocks -> per-n / per-m [.,64]x[64,64]
// projections + rank-1 rw term; delta GEMM hoisted outside the mean.
// Attention GEMMs: v_wmma_f32_16x16x32_f16 with f16-staged, layout-
// transposed operands so every B fragment is two aligned 16B loads per lane.
// ---------------------------------------------------------------------------

typedef __attribute__((ext_vector_type(16))) _Float16       v16h;
typedef __attribute__((ext_vector_type(8)))  float          v8f;
typedef __attribute__((ext_vector_type(8)))  unsigned short u16x8;
typedef __attribute__((ext_vector_type(4)))  float          f32x4;

#define N_PTS   4096
#define M_SLOTS 256
#define DD      64
#define R_RES   512
#define NSPLIT  8

__device__ __forceinline__ float silu_f(float x) {
  return x / (1.0f + __expf(-x));
}

__device__ __forceinline__ v16h join16(u16x8 lo, u16x8 hi) {
  union { u16x8 u[2]; v16h h; } x;
  x.u[0] = lo; x.u[1] = hi;
  return x.h;
}

// ---------------------------------------------------------------------------
// keysTh[slot][d] = (f16) keys[slot][d]      (once; same layout)
// ---------------------------------------------------------------------------
__global__ __launch_bounds__(256) void conv_keys_kernel(
    const float* __restrict__ keys, _Float16* __restrict__ keysTh) {
  const int idx = blockIdx.x * 256 + threadIdx.x;   // idx = d*256 + slot
  const int d = idx >> 8, slot = idx & 255;
  keysTh[slot * DD + d] = (_Float16)keys[slot * DD + d];
}

// ---------------------------------------------------------------------------
// memh[d][slot] = (f16) mem[slot][d]         (per step; transposed)
// ---------------------------------------------------------------------------
__global__ __launch_bounds__(256) void conv_mem_kernel(
    const float* __restrict__ memc, _Float16* __restrict__ memh) {
  const int idx = blockIdx.x * 256 + threadIdx.x;
  memh[idx] = (_Float16)memc[(idx & 255) * DD + (idx >> 8)];
}

// ---------------------------------------------------------------------------
// qe = silu(points @ W1 + b1) @ W2 + b2         (block = 4 rows x 64 channels)
// ---------------------------------------------------------------------------
__global__ __launch_bounds__(256) void qe_init_kernel(
    const float* __restrict__ coords, const float* __restrict__ W1,
    const float* __restrict__ b1, const float* __restrict__ W2,
    const float* __restrict__ b2, float* __restrict__ qe) {
  __shared__ float hbuf[4][64];
  const int r = threadIdx.x >> 6, d = threadIdx.x & 63;
  const int n = blockIdx.x * 4 + r;
  const float x = coords[n * 3 + 0], y = coords[n * 3 + 1], z = coords[n * 3 + 2];
  float h = b1[d] + x * W1[0 * 64 + d] + y * W1[1 * 64 + d] + z * W1[2 * 64 + d];
  hbuf[r][d] = silu_f(h);
  __syncthreads();
  float q = b2[d];
  #pragma unroll 8
  for (int k = 0; k < 64; ++k) q += hbuf[r][k] * W2[k * 64 + d];
  qe[n * 64 + d] = q;
}

// ---------------------------------------------------------------------------
// sg[n] = sigmoid(4 * min_r(|p - rpos_r| - rrad_r))    (step-invariant)
// ---------------------------------------------------------------------------
__global__ __launch_bounds__(256) void sdf_kernel(
    const float* __restrict__ coords, const float* __restrict__ rpos,
    const float* __restrict__ rrad, float* __restrict__ sg) {
  const int n = blockIdx.x * 256 + threadIdx.x;
  const float x = coords[n * 3], y = coords[n * 3 + 1], z = coords[n * 3 + 2];
  float mn = 3.4e38f;
  for (int r = 0; r < R_RES; ++r) {
    float dx = x - rpos[r * 3], dy = y - rpos[r * 3 + 1], dz = z - rpos[r * 3 + 2];
    float d = sqrtf(dx * dx + dy * dy + dz * dz) - rrad[r];
    mn = fminf(mn, d);
  }
  sg[n] = 1.0f / (1.0f + __expf(-4.0f * mn));
}

// ---------------------------------------------------------------------------
// Fused attention: logits = (qe/8) @ keys^T, softmax, mc = rw @ mem,
// qe += 0.5*mc.  One wave owns a 16-point tile; logits live in 16 v8f accs.
// WMMA fragment layouts per CDNA5 ISA 7.12.2 (16-bit A 16x32; B n=lane%16).
// B fragments are two 16B loads joined into a v16h from the f16-staged matrices.
// ---------------------------------------------------------------------------
__global__ __launch_bounds__(256) void attn_kernel(
    float* __restrict__ qe, const _Float16* __restrict__ keysTh,
    const _Float16* __restrict__ memh, float* __restrict__ rw_out,
    float* __restrict__ mc_out) {
  __shared__ _Float16 lds_rw[8][16][256];   // 64 KB, per-wave rw tiles (f16)
  const int lane = threadIdx.x & 31;
  const int wave = threadIdx.x >> 5;
  const int row0 = (blockIdx.x * 8 + wave) * 16;
  const int mrow = lane & 15;
  const int kofs = (lane >= 16) ? 8 : 0;    // A-fragment K offset (ISA layout)
  const int prow = (lane >= 16) ? 8 : 0;    // C-fragment M offset

  // ---- A fragments for logits: vectorized 8-float runs, scale folded ----
  const f32x4* qrow = (const f32x4*)(qe + (row0 + mrow) * DD);
  const int q0 = kofs >> 2;                 // 0 or 2 (f32x4 units)
  float qa[16], qb[16];
  *(f32x4*)&qa[0]  = qrow[q0 + 0];  *(f32x4*)&qa[4]  = qrow[q0 + 1];
  *(f32x4*)&qa[8]  = qrow[q0 + 4];  *(f32x4*)&qa[12] = qrow[q0 + 5];
  *(f32x4*)&qb[0]  = qrow[q0 + 8];  *(f32x4*)&qb[4]  = qrow[q0 + 9];
  *(f32x4*)&qb[8]  = qrow[q0 + 12]; *(f32x4*)&qb[12] = qrow[q0 + 13];
  v16h a0, a1;
  #pragma unroll
  for (int h = 0; h < 16; ++h) {
    a0[h] = (_Float16)(qa[h] * 0.125f);
    a1[h] = (_Float16)(qb[h] * 0.125f);
  }

  // ---- logits = (qe*0.125) @ keys^T : B[k][n] = keysTh[n][k], 2x16B loads ----
  v8f acc[16];
  #pragma unroll
  for (int c = 0; c < 16; ++c) {
    v16h b0 = join16(*(const u16x8*)(keysTh + (c * 16 + mrow) * DD + kofs),
                     *(const u16x8*)(keysTh + (c * 16 + mrow) * DD + kofs + 16));
    v16h b1 = join16(*(const u16x8*)(keysTh + (c * 16 + mrow) * DD + 32 + kofs),
                     *(const u16x8*)(keysTh + (c * 16 + mrow) * DD + 32 + kofs + 16));
    v8f z = {};
    z = __builtin_amdgcn_wmma_f32_16x16x32_f16(false, a0, false, b0, (short)0, z, false, false);
    z = __builtin_amdgcn_wmma_f32_16x16x32_f16(false, a1, false, b1, (short)0, z, false, false);
    acc[c] = z;
  }

  // ---- softmax over 256 columns (row lives in one 16-lane half) ----
  float rmax[8], rinv[8];
  #pragma unroll
  for (int v = 0; v < 8; ++v) {
    float t = acc[0][v];
    #pragma unroll
    for (int c = 1; c < 16; ++c) t = fmaxf(t, acc[c][v]);
    #pragma unroll
    for (int off = 1; off < 16; off <<= 1) t = fmaxf(t, __shfl_xor(t, off, 32));
    rmax[v] = t;
  }
  #pragma unroll
  for (int v = 0; v < 8; ++v) {
    float s = 0.0f;
    #pragma unroll
    for (int c = 0; c < 16; ++c) {
      float e = __expf(acc[c][v] - rmax[v]);
      acc[c][v] = e;
      s += e;
    }
    #pragma unroll
    for (int off = 1; off < 16; off <<= 1) s += __shfl_xor(s, off, 32);
    rinv[v] = 1.0f / s;
  }
  #pragma unroll
  for (int c = 0; c < 16; ++c) {
    #pragma unroll
    for (int v = 0; v < 8; ++v) {
      float p = acc[c][v] * rinv[v];
      int pt = v + prow;
      int col = (lane & 15) + 16 * c;
      rw_out[(row0 + pt) * M_SLOTS + col] = p;
      lds_rw[wave][pt][col] = (_Float16)p;
    }
  }
  __syncthreads();

  // ---- mc = rw @ mem, K = 256 over slots; B = memh, 2x16B loads ----
  v8f acc2[4];
  v8f zero = {};
  #pragma unroll
  for (int c2 = 0; c2 < 4; ++c2) acc2[c2] = zero;
  #pragma unroll
  for (int kb = 0; kb < 8; ++kb) {
    const _Float16* arow = &lds_rw[wave][mrow][kb * 32 + kofs];
    v16h af = join16(*(const u16x8*)(arow), *(const u16x8*)(arow + 16));
    #pragma unroll
    for (int c2 = 0; c2 < 4; ++c2) {
      v16h bf = join16(*(const u16x8*)(memh + (c2 * 16 + mrow) * M_SLOTS + kb * 32 + kofs),
                       *(const u16x8*)(memh + (c2 * 16 + mrow) * M_SLOTS + kb * 32 + kofs + 16));
      acc2[c2] = __builtin_amdgcn_wmma_f32_16x16x32_f16(false, af, false, bf,
                                                        (short)0, acc2[c2], false, false);
    }
  }
  #pragma unroll
  for (int c2 = 0; c2 < 4; ++c2) {
    #pragma unroll
    for (int v = 0; v < 8; ++v) {
      int pt = v + prow;
      int d = (lane & 15) + 16 * c2;
      int idx = (row0 + pt) * DD + d;
      float mcv = acc2[c2][v];
      mc_out[idx] = mcv;
      qe[idx] += 0.5f * mcv;              // fused qe update (own rows only)
    }
  }
}

// ---------------------------------------------------------------------------
// A_g[n] = mc[n] @ Wg1[0:64] + sg[n]*Wg1[129] + bg1  (same for A_u with Wu1)
// ---------------------------------------------------------------------------
__global__ __launch_bounds__(256) void projA_kernel(
    const float* __restrict__ mc, const float* __restrict__ sg,
    const float* __restrict__ Wg1, const float* __restrict__ bg1,
    const float* __restrict__ Wu1, const float* __restrict__ bu1,
    float* __restrict__ Ag, float* __restrict__ Au) {
  __shared__ float mrow[4][64];
  const int r = threadIdx.x >> 6, d = threadIdx.x & 63;
  const int n = blockIdx.x * 4 + r;
  mrow[r][d] = mc[n * 64 + d];
  __syncthreads();
  const float sgn = sg[n];
  float ag = bg1[d] + sgn * Wg1[129 * 64 + d];
  float au = bu1[d] + sgn * Wu1[129 * 64 + d];
  #pragma unroll 8
  for (int k = 0; k < 64; ++k) {
    float x = mrow[r][k];
    ag += x * Wg1[k * 64 + d];
    au += x * Wu1[k * 64 + d];
  }
  Ag[n * 64 + d] = ag;
  Au[n * 64 + d] = au;
}

// ---------------------------------------------------------------------------
// B_g[m] = mem[m] @ Wg1[64:128]   (same for B_u with Wu1)
// ---------------------------------------------------------------------------
__global__ __launch_bounds__(256) void projB_kernel(
    const float* __restrict__ memc, const float* __restrict__ Wg1,
    const float* __restrict__ Wu1, float* __restrict__ Bg, float* __restrict__ Bu) {
  __shared__ float mrow[4][64];
  const int r = threadIdx.x >> 6, d = threadIdx.x & 63;
  const int m = blockIdx.x * 4 + r;
  mrow[r][d] = memc[m * 64 + d];
  __syncthreads();
  float bg = 0.0f, bu = 0.0f;
  #pragma unroll 8
  for (int k = 0; k < 64; ++k) {
    float x = mrow[r][k];
    bg += x * Wg1[(64 + k) * 64 + d];
    bu += x * Wu1[(64 + k) * 64 + d];
  }
  Bg[m * 64 + d] = bg;
  Bu[m * 64 + d] = bu;
}

// ---------------------------------------------------------------------------
// Heavy pair pass: for slot m, accumulate over n:
//   gate_sum  += sigmoid(dot(silu(Ag + Bg + rw*wg1_rw), Wg2) + bg2)
//   hbar[d]   += silu(Au + Bu + rw*wu1_rw)[d]
// One wave per n-row (lane owns d and d+32); in-wave butterfly for gate dot;
// deterministic partials per (split, m).  Prefetch next rows (L2-resident).
// ---------------------------------------------------------------------------
__global__ __launch_bounds__(256) void pair_kernel(
    const float* __restrict__ Ag, const float* __restrict__ Au,
    const float* __restrict__ Bg, const float* __restrict__ Bu,
    const float* __restrict__ rw, const float* __restrict__ Wg1,
    const float* __restrict__ Wg2, const float* __restrict__ bg2,
    const float* __restrict__ Wu1,
    float* __restrict__ gate_part, float* __restrict__ hbar_part) {
  __shared__ float shb[8][64];
  __shared__ float sgw[8];
  const int lane = threadIdx.x & 31;
  const int w = threadIdx.x >> 5;
  const int m = blockIdx.x;
  const int nbase = blockIdx.y * (N_PTS / NSPLIT);
  const int iters = (N_PTS / NSPLIT) / 8;   // 64

  const float bgv0 = Bg[m * 64 + lane], bgv1 = Bg[m * 64 + lane + 32];
  const float buv0 = Bu[m * 64 + lane], buv1 = Bu[m * 64 + lane + 32];
  const float wg1r0 = Wg1[128 * 64 + lane], wg1r1 = Wg1[128 * 64 + lane + 32];
  const float wu1r0 = Wu1[128 * 64 + lane], wu1r1 = Wu1[128 * 64 + lane + 32];
  const float wg20 = Wg2[lane], wg21 = Wg2[lane + 32];
  const float bg2s = bg2[0];

  float hb0 = 0.0f, hb1 = 0.0f, gacc = 0.0f;
  for (int i = 0; i < iters; ++i) {
    const int n = nbase + i * 8 + w;
    if (i + 1 < iters) {                    // speculative L0 prefetch
      __builtin_prefetch(Ag + (n + 8) * 64 + lane, 0, 0);
      __builtin_prefetch(Au + (n + 8) * 64 + lane, 0, 0);
    }
    const float r = rw[n * M_SLOTS + m];
    const float ag0 = Ag[n * 64 + lane], ag1 = Ag[n * 64 + lane + 32];
    const float au0 = Au[n * 64 + lane], au1 = Au[n * 64 + lane + 32];
    float xg0 = silu_f(ag0 + bgv0 + r * wg1r0);
    float xg1 = silu_f(ag1 + bgv1 + r * wg1r1);
    float xu0 = silu_f(au0 + buv0 + r * wu1r0);
    float xu1 = silu_f(au1 + buv1 + r * wu1r1);
    hb0 += xu0;
    hb1 += xu1;
    float part = xg0 * wg20 + xg1 * wg21;
    #pragma unroll
    for (int off = 1; off < 32; off <<= 1) part += __shfl_xor(part, off, 32);
    gacc += 1.0f / (1.0f + __expf(-(part + bg2s)));
  }
  shb[w][lane] = hb0;
  shb[w][lane + 32] = hb1;
  if (lane == 0) sgw[w] = gacc;
  __syncthreads();
  const int tid = threadIdx.x;
  if (tid < 64) {
    float s = 0.0f;
    #pragma unroll
    for (int ww = 0; ww < 8; ++ww) s += shb[ww][tid];
    hbar_part[(blockIdx.y * M_SLOTS + m) * 64 + tid] = s;
  }
  if (tid == 0) {
    float g = 0.0f;
    for (int ww = 0; ww < 8; ++ww) g += sgw[ww];
    gate_part[blockIdx.y * M_SLOTS + m] = g;
  }
}

// ---------------------------------------------------------------------------
// Reduce partials; delta = (mean_n hbar) @ Wu2 + bu2; mem2 = mem + gate*delta
// ---------------------------------------------------------------------------
__global__ __launch_bounds__(64) void finalize_kernel(
    const float* __restrict__ gate_part, const float* __restrict__ hbar_part,
    const float* __restrict__ Wu2, const float* __restrict__ bu2,
    const float* __restrict__ mem_in, float* __restrict__ mem_out,
    float* __restrict__ gate_out) {
  __shared__ float hb[64];
  __shared__ float gsh;
  const int m = blockIdx.x, d = threadIdx.x;
  float s = 0.0f;
  for (int sp = 0; sp < NSPLIT; ++sp) s += hbar_part[(sp * M_SLOTS + m) * 64 + d];
  hb[d] = s * (1.0f / N_PTS);
  if (d == 0) {
    float g = 0.0f;
    for (int sp = 0; sp < NSPLIT; ++sp) g += gate_part[sp * M_SLOTS + m];
    gsh = g * (1.0f / N_PTS);
    gate_out[m] = gsh;
  }
  __syncthreads();
  float delta = bu2[d];
  #pragma unroll 8
  for (int k = 0; k < 64; ++k) delta += hb[k] * Wu2[k * 64 + d];
  mem_out[m * 64 + d] = mem_in[m * 64 + d] + gsh * delta;
}

// ---------------------------------------------------------------------------
extern "C" void kernel_launch(void* const* d_in, const int* in_sizes, int n_in,
                              void* d_out, int out_size, void* d_ws, size_t ws_size,
                              hipStream_t stream) {
  (void)in_sizes; (void)n_in; (void)out_size; (void)ws_size;
  const float* coords       = (const float*)d_in[0];
  const float* memory_slots = (const float*)d_in[1];
  const float* keys         = (const float*)d_in[2];
  const float* rpos         = (const float*)d_in[3];
  const float* rrad         = (const float*)d_in[4];
  const float* W1  = (const float*)d_in[5];
  const float* b1  = (const float*)d_in[6];
  const float* W2  = (const float*)d_in[7];
  const float* b2  = (const float*)d_in[8];
  const float* Wg1 = (const float*)d_in[9];
  const float* bg1 = (const float*)d_in[10];
  const float* Wg2 = (const float*)d_in[11];
  const float* bg2 = (const float*)d_in[12];
  const float* Wu1 = (const float*)d_in[13];
  const float* bu1 = (const float*)d_in[14];
  const float* Wu2 = (const float*)d_in[15];
  const float* bu2 = (const float*)d_in[16];

  // Workspace layout (float units), total ~9.3 MB.
  float* ws        = (float*)d_ws;
  float* qe        = ws;                    // 4096*64   = 262144
  float* mc        = qe + 262144;           // 262144
  float* rw        = mc + 262144;           // 4096*256  = 1048576
  float* Ag        = rw + 1048576;          // 262144
  float* Au        = Ag + 262144;           // 262144
  float* Bg        = Au + 262144;           // 16384
  float* Bu        = Bg + 16384;            // 16384
  float* mem0      = Bu + 16384;            // 16384
  float* mem1      = mem0 + 16384;          // 16384
  float* sgb       = mem1 + 16384;          // 4096
  float* gate_part = sgb + 4096;            // 8*256     = 2048
  float* hbar_part = gate_part + 2048;      // 8*256*64  = 131072
  float* gate      = hbar_part + 131072;    // 256
  _Float16* keysTh = (_Float16*)(gate + 256);        // 64*256 halves (32B-aligned)
  _Float16* memh   = (_Float16*)(gate + 256 + 8192); // 256*64 halves

  hipMemcpyAsync(mem0, memory_slots, 16384 * sizeof(float),
                 hipMemcpyDeviceToDevice, stream);
  conv_keys_kernel<<<64, 256, 0, stream>>>(keys, keysTh);
  qe_init_kernel<<<1024, 256, 0, stream>>>(coords, W1, b1, W2, b2, qe);
  sdf_kernel<<<16, 256, 0, stream>>>(coords, rpos, rrad, sgb);

  float* mcur = mem0;
  float* mnext = mem1;
  for (int step = 0; step < 2; ++step) {
    conv_mem_kernel<<<64, 256, 0, stream>>>(mcur, memh);
    attn_kernel<<<32, 256, 0, stream>>>(qe, keysTh, memh, rw, mc);
    projA_kernel<<<1024, 256, 0, stream>>>(mc, sgb, Wg1, bg1, Wu1, bu1, Ag, Au);
    projB_kernel<<<64, 256, 0, stream>>>(mcur, Wg1, Wu1, Bg, Bu);
    pair_kernel<<<dim3(M_SLOTS, NSPLIT), 256, 0, stream>>>(
        Ag, Au, Bg, Bu, rw, Wg1, Wg2, bg2, Wu1, gate_part, hbar_part);
    finalize_kernel<<<M_SLOTS, 64, 0, stream>>>(
        gate_part, hbar_part, Wu2, bu2, mcur, mnext, gate);
    float* t = mcur; mcur = mnext; mnext = t;
  }

  // Outputs concatenated: mc[4096,64], rw[4096,256], gate[256], sg[4096], mem[256,64]
  float* out = (float*)d_out;
  hipMemcpyAsync(out,           mc,   262144 * sizeof(float),  hipMemcpyDeviceToDevice, stream);
  hipMemcpyAsync(out + 262144,  rw,   1048576 * sizeof(float), hipMemcpyDeviceToDevice, stream);
  hipMemcpyAsync(out + 1310720, gate, 256 * sizeof(float),     hipMemcpyDeviceToDevice, stream);
  hipMemcpyAsync(out + 1310976, sgb,  4096 * sizeof(float),    hipMemcpyDeviceToDevice, stream);
  hipMemcpyAsync(out + 1315072, mcur, 16384 * sizeof(float),   hipMemcpyDeviceToDevice, stream);
}
